// JointAttention_81570018886116
// MI455X (gfx1250) — matrix operand order repaired and checked by reading, not compiled, not run on hardware
//
#include <hip/hip_runtime.h>
#include <hip/hip_bf16.h>
#include <cstdint>

#define B_   2
#define S_   2048
#define DIM_ 2048
#define H_   16
#define HKV_ 8
#define HD_  128

typedef __attribute__((ext_vector_type(16))) __bf16 bf16x16;
typedef __attribute__((ext_vector_type(8)))  float  f32x8;

#ifndef __has_builtin
#define __has_builtin(x) 0
#endif
#if __has_builtin(__builtin_amdgcn_global_load_async_to_lds_b128) && \
    __has_builtin(__builtin_amdgcn_s_wait_asynccnt)
#define HAVE_ASYNC 1
#else
#define HAVE_ASYNC 0
#endif

#if HAVE_ASYNC
typedef int v4i_ __attribute__((ext_vector_type(4)));
typedef __attribute__((address_space(1))) v4i_* gv4p;   // global int4*
typedef __attribute__((address_space(3))) v4i_* lv4p;   // LDS int4*
// Copy 64 contiguous bytes global -> LDS via the CDNA5 async path (ASYNCcnt).
static __device__ __forceinline__ void cp64(const __bf16* g, __bf16* l) {
    gv4p gp = (gv4p)(void*)const_cast<__bf16*>(g);
    lv4p lp = (lv4p)(void*)l;
    __builtin_amdgcn_global_load_async_to_lds_b128(gp, lp, 0, 0);
    __builtin_amdgcn_global_load_async_to_lds_b128(gp, lp, 16, 0);
    __builtin_amdgcn_global_load_async_to_lds_b128(gp, lp, 32, 0);
    __builtin_amdgcn_global_load_async_to_lds_b128(gp, lp, 48, 0);
}
#define ASYNC_WAIT() __builtin_amdgcn_s_wait_asynccnt(0)
#else
static __device__ __forceinline__ void cp64(const __bf16* g, __bf16* l) {
    uint4 t0 = ((const uint4*)g)[0];
    uint4 t1 = ((const uint4*)g)[1];
    uint4 t2 = ((const uint4*)g)[2];
    uint4 t3 = ((const uint4*)g)[3];
    ((uint4*)l)[0] = t0; ((uint4*)l)[1] = t1;
    ((uint4*)l)[2] = t2; ((uint4*)l)[3] = t3;
}
#define ASYNC_WAIT() ((void)0)
#endif

static __device__ __forceinline__ f32x8 wmma_bf16(bf16x16 a, bf16x16 b, f32x8 c) {
    return __builtin_amdgcn_wmma_f32_16x16x32_bf16(false, a, false, b, (short)0, c, false, false);
}

// A-operand fragment: 16x32 tile from row-major [rows, K] region.
static __device__ __forceinline__ bf16x16 load_frag_a(const __bf16* base, int pitch, int lane) {
    int r    = lane & 15;
    int koff = (lane >> 4) << 3;            // 0 or 8
    const __bf16* p = base + r * pitch + koff;
    union { bf16x16 v; uint4 q[2]; } u;
    u.q[0] = *(const uint4*)(p);
    u.q[1] = *(const uint4*)(p + 16);
    return u.v;
}

// B-operand fragment: 32x16 tile from row-major [N, K] region (NT gemm).
static __device__ __forceinline__ bf16x16 load_frag_b(const __bf16* base, int pitch, int lane) {
    int n    = lane & 15;
    int koff = (lane >> 4) << 3;            // 0 or 8
    const __bf16* p = base + n * pitch + koff;
    union { bf16x16 v; uint4 q[2]; } u;
    u.q[0] = *(const uint4*)(p);
    u.q[1] = *(const uint4*)(p + 16);
    return u.v;
}

// ---------------- fp32 -> bf16 bulk convert (one-shot pre-pass) ----------------------
__global__ __launch_bounds__(256) void cvt_bf16_kernel(const float* __restrict__ in,
                                                       __bf16* __restrict__ out, int n4) {
    int i = blockIdx.x * 256 + threadIdx.x;
    const int stride = gridDim.x * 256;
    for (; i < n4; i += stride) {
        float4 v = ((const float4*)in)[i];
        union { __bf16 h[4]; uint2 u; } st;
        st.h[0] = (__bf16)v.x; st.h[1] = (__bf16)v.y;
        st.h[2] = (__bf16)v.z; st.h[3] = (__bf16)v.w;
        ((uint2*)out)[i] = st.u;
    }
}

// ---------------- GEMM: C[M,N] = A[M,K] @ Bw[N,K]^T  (bf16 in, fp32 out) -------------
// 128x128 block tile, double-buffered LDS, async global->LDS fills.
__global__ __launch_bounds__(256) void gemm_nt_bf(const __bf16* __restrict__ A,
                                                  const __bf16* __restrict__ Bw,
                                                  float* __restrict__ C,
                                                  int M, int N, int K) {
    const int TP = 40;                       // 32 + 8 pad, 80B rows
    __shared__ __align__(16) __bf16 As[2][128 * 40];
    __shared__ __align__(16) __bf16 Bs[2][128 * 40];

    const int tid  = threadIdx.x;
    const int lane = tid & 31;
    const int wave = tid >> 5;
    const int wy   = wave >> 2;              // 0..1 -> 64 rows each
    const int wx   = wave & 3;               // 0..3 -> 32 cols each
    const int bm   = blockIdx.y * 128;
    const int bn   = blockIdx.x * 128;

    f32x8 acc[4][2] = {};

    // loader: threads 0..127 own one A row, 128..255 own one B row (64B per k-tile)
    const bool isA = tid < 128;
    const int  lrow = tid & 127;
    const __bf16* gsrc = isA ? (A  + (size_t)(bm + lrow) * K)
                             : (Bw + (size_t)(bn + lrow) * K);

    __bf16* myl[2] = { (isA ? As[0] : Bs[0]) + lrow * TP,
                       (isA ? As[1] : Bs[1]) + lrow * TP };

    cp64(gsrc, myl[0]);                      // prologue fill buf0
    ASYNC_WAIT();
    __syncthreads();

    int cur = 0;
    for (int kb = 0; kb < K; kb += 32) {
        if (kb + 32 < K) cp64(gsrc + kb + 32, myl[cur ^ 1]);

        const __bf16* cA = As[cur];
        const __bf16* cB = Bs[cur];
        bf16x16 af[4];
#pragma unroll
        for (int i = 0; i < 4; ++i)
            af[i] = load_frag_a(cA + (wy * 64 + i * 16) * TP, TP, lane);
#pragma unroll
        for (int j = 0; j < 2; ++j) {
            bf16x16 bf = load_frag_b(cB + (wx * 32 + j * 16) * TP, TP, lane);
#pragma unroll
            for (int i = 0; i < 4; ++i)
                acc[i][j] = wmma_bf16(af[i], bf, acc[i][j]);
        }

        ASYNC_WAIT();
        __syncthreads();
        cur ^= 1;
    }

    const int col0  = bn + wx * 32 + (lane & 15);
    const int rbase = bm + wy * 64 + ((lane >> 4) << 3);
#pragma unroll
    for (int i = 0; i < 4; ++i)
#pragma unroll
        for (int j = 0; j < 2; ++j) {
            float* cp = C + (size_t)(rbase + i * 16) * N + col0 + j * 16;
#pragma unroll
            for (int e = 0; e < 8; ++e) cp[(size_t)e * N] = acc[i][j][e];
        }
}

// ---------------- Prep: rmsnorm + rope + bf16 pack, V transposed ---------------------
__global__ __launch_bounds__(256) void prep_kernel(const float* __restrict__ qkv,
                                                   const float* __restrict__ cis,
                                                   const float* __restrict__ qnw,
                                                   const float* __restrict__ knw,
                                                   __bf16* __restrict__ Qb,
                                                   __bf16* __restrict__ Kb,
                                                   __bf16* __restrict__ Vtb) {
    const int row  = blockIdx.x;             // b*S + s
    const int b    = row / S_;
    const int s    = row % S_;
    const int tid  = threadIdx.x;
    const int lane = tid & 31;
    const int wave = tid >> 5;
    const float* cbase = cis + (size_t)row * 64 * 2;   // [64 pairs][cos,sin]

    for (int t = 0; t < 4; ++t) {
        const int slot = wave * 4 + t;       // 0..15 Q, 16..23 K, 24..31 V
        const float* src = qkv + (size_t)row * 4096 + slot * 128 + lane * 4;
        float4 x = *(const float4*)src;
        if (slot < 24) {
            float ss = x.x * x.x + x.y * x.y + x.z * x.z + x.w * x.w;
#pragma unroll
            for (int off = 1; off < 32; off <<= 1) ss += __shfl_xor(ss, off, 32);
            const float r = rsqrtf(ss * (1.0f / 128.0f) + 1e-6f);
            const float* nw = (slot < 16 ? qnw : knw) + lane * 4;
            float y0 = x.x * r * nw[0], y1 = x.y * r * nw[1];
            float y2 = x.z * r * nw[2], y3 = x.w * r * nw[3];
            const int j0 = lane * 2;
            float c0 = cbase[j0 * 2 + 0], s0 = cbase[j0 * 2 + 1];
            float c1 = cbase[j0 * 2 + 2], s1 = cbase[j0 * 2 + 3];
            union { __bf16 hv[4]; uint2 u; } st;
            st.hv[0] = (__bf16)(y0 * c0 - y1 * s0);
            st.hv[1] = (__bf16)(y0 * s0 + y1 * c0);
            st.hv[2] = (__bf16)(y2 * c1 - y3 * s1);
            st.hv[3] = (__bf16)(y2 * s1 + y3 * c1);
            if (slot < 16) {
                __bf16* dst = Qb + ((size_t)(b * H_ + slot) * S_ + s) * HD_ + lane * 4;
                *(uint2*)dst = st.u;
            } else {
                __bf16* dst = Kb + ((size_t)(b * HKV_ + (slot - 16)) * S_ + s) * HD_ + lane * 4;
                *(uint2*)dst = st.u;
            }
        } else {
            const int kh = slot - 24;        // V: transpose to [bh][hd][s]
            size_t base = ((size_t)(b * HKV_ + kh) * HD_ + lane * 4) * S_ + s;
            Vtb[base + 0 * S_] = (__bf16)x.x;
            Vtb[base + 1 * S_] = (__bf16)x.y;
            Vtb[base + 2 * S_] = (__bf16)x.z;
            Vtb[base + 3 * S_] = (__bf16)x.w;
        }
    }
}

// ---------------- Flash attention: per (b,h) head, 128 q rows per block --------------
__global__ __launch_bounds__(256) void attn_kernel(const __bf16* __restrict__ Qb,
                                                   const __bf16* __restrict__ Kb,
                                                   const __bf16* __restrict__ Vt,
                                                   const unsigned char* __restrict__ maskp,
                                                   __bf16* __restrict__ Ob) {
    __shared__ __align__(16) __bf16 Ks[64 * 136];      // [key][hd], pitch 136
    __shared__ __align__(16) __bf16 Vs[128 * 72];      // [hd][key], pitch 72
    __shared__ __align__(16) __bf16 Ps[8][16 * 72];    // per-wave P, pitch 72

    const int tid  = threadIdx.x;
    const int lane = tid & 31;
    const int wave = tid >> 5;
    const int bh   = blockIdx.y;             // b*H + h
    const int b    = bh / H_;
    const int h    = bh % H_;
    const int kh   = h >> 1;                 // GQA: 2 q heads per kv head
    const int q0   = blockIdx.x * 128 + wave * 16;

    const __bf16* qptr = Qb + ((size_t)bh * S_ + q0) * HD_;
    bf16x16 qf[4];
#pragma unroll
    for (int kk = 0; kk < 4; ++kk) qf[kk] = load_frag_a(qptr + kk * 32, HD_, lane);

    f32x8 o[8] = {};
    float m[8], l[8];
#pragma unroll
    for (int e = 0; e < 8; ++e) { m[e] = -3.0e38f; l[e] = 0.0f; }

    const __bf16* kbase = Kb + (size_t)(b * HKV_ + kh) * S_ * HD_;
    const __bf16* vbase = Vt + (size_t)(b * HKV_ + kh) * HD_ * S_;
    const unsigned char* mbase = maskp + (size_t)b * S_;
    const float sc = 0.08838834764831845f;   // 1/sqrt(128)

    // per-thread tile-copy assignments (64B each, contiguous both sides)
    const int krow = tid >> 2, kcol = (tid & 3) * 32;
    const int vrow = tid >> 1, vcol = (tid & 1) * 32;

    for (int kb0 = 0; kb0 < S_; kb0 += 64) {
        cp64(kbase + (size_t)(kb0 + krow) * HD_ + kcol, Ks + krow * 136 + kcol);
        cp64(vbase + (size_t)vrow * S_ + kb0 + vcol,    Vs + vrow * 72  + vcol);
        ASYNC_WAIT();
        __syncthreads();

        // scores: 16 q rows x 64 keys
        f32x8 s[4] = {};
#pragma unroll
        for (int kk = 0; kk < 4; ++kk)
#pragma unroll
            for (int j = 0; j < 4; ++j) {
                bf16x16 bf = load_frag_b(Ks + (j * 16) * 136 + kk * 32, 136, lane);
                s[j] = wmma_bf16(qf[kk], bf, s[j]);
            }

        // scale + mask
#pragma unroll
        for (int j = 0; j < 4; ++j) {
            const bool keep = mbase[kb0 + j * 16 + (lane & 15)] != 0;
#pragma unroll
            for (int e = 0; e < 8; ++e) {
                float v = s[j][e] * sc;
                s[j][e] = keep ? v : -1.0e9f;
            }
        }

        // online softmax stats (rows on element axis; reduce 16-lane halves)
        float mn[8], al[8], rs[8];
#pragma unroll
        for (int e = 0; e < 8; ++e) {
            float v = fmaxf(fmaxf(s[0][e], s[1][e]), fmaxf(s[2][e], s[3][e]));
#pragma unroll
            for (int off = 1; off < 16; off <<= 1) v = fmaxf(v, __shfl_xor(v, off, 32));
            mn[e] = fmaxf(m[e], v);
            al[e] = __expf(m[e] - mn[e]);
            m[e]  = mn[e];
            rs[e] = 0.0f;
        }

        // P = exp(s - m): transpose via per-wave LDS into A-fragment layout
        __bf16* pw = &Ps[wave][0];
        const int prow = (lane >> 4) << 3;
        const int pcol = lane & 15;
#pragma unroll
        for (int j = 0; j < 4; ++j)
#pragma unroll
            for (int e = 0; e < 8; ++e) {
                float p = __expf(s[j][e] - mn[e]);
                rs[e] += p;
                pw[(e + prow) * 72 + j * 16 + pcol] = (__bf16)p;
            }
#pragma unroll
        for (int e = 0; e < 8; ++e) {
            float v = rs[e];
#pragma unroll
            for (int off = 1; off < 16; off <<= 1) v += __shfl_xor(v, off, 32);
            l[e] = l[e] * al[e] + v;
        }
#pragma unroll
        for (int i = 0; i < 8; ++i)
#pragma unroll
            for (int e = 0; e < 8; ++e) o[i][e] *= al[e];

        // O += P @ V
#pragma unroll
        for (int ks = 0; ks < 2; ++ks) {
            bf16x16 af = load_frag_a(pw + ks * 32, 72, lane);
#pragma unroll
            for (int i = 0; i < 8; ++i) {
                bf16x16 bf = load_frag_b(Vs + (i * 16) * 72 + ks * 32, 72, lane);
                o[i] = wmma_bf16(af, bf, o[i]);
            }
        }
        __syncthreads();
    }

    // normalize + store bf16 [B,S,H,HD]
    const int rbase = q0 + ((lane >> 4) << 3);
    const int cbase2 = lane & 15;
#pragma unroll
    for (int i = 0; i < 8; ++i)
#pragma unroll
        for (int e = 0; e < 8; ++e) {
            float v = o[i][e] / l[e];
            size_t idx = (((size_t)b * S_ + rbase + e) * H_ + h) * HD_ + i * 16 + cbase2;
            Ob[idx] = (__bf16)v;
        }
}

extern "C" void kernel_launch(void* const* d_in, const int* in_sizes, int n_in,
                              void* d_out, int out_size, void* d_ws, size_t ws_size,
                              hipStream_t stream) {
    const float*         x     = (const float*)d_in[0];
    const unsigned char* xmask = (const unsigned char*)d_in[1];
    const float*         cis   = (const float*)d_in[2];
    const float*         w_qkv = (const float*)d_in[3];
    const float*         w_out = (const float*)d_in[4];
    const float*         qnw   = (const float*)d_in[5];
    const float*         knw   = (const float*)d_in[6];
    float*               out   = (float*)d_out;

    char* ws = (char*)d_ws;
    const size_t MB = 1024 * 1024;
    float*  qkv    = (float*)ws;                         // 64 MB fp32 [4096,4096]
    __bf16* Qb     = (__bf16*)(ws + 64  * MB);           // 16 MB
    __bf16* Kb     = (__bf16*)(ws + 80  * MB);           //  8 MB
    __bf16* Vt     = (__bf16*)(ws + 88  * MB);           //  8 MB
    __bf16* Ob     = (__bf16*)(ws + 96  * MB);           // 16 MB
    __bf16* xbf    = (__bf16*)(ws + 112 * MB);           // 16 MB
    __bf16* wqkvbf = (__bf16*)(ws + 128 * MB);           // 16 MB
    __bf16* woutbf = (__bf16*)(ws + 144 * MB);           //  8 MB

    dim3 blk(256);
    cvt_bf16_kernel<<<dim3(2048), blk, 0, stream>>>(x,     xbf,    (4096 * 2048) / 4);
    cvt_bf16_kernel<<<dim3(2048), blk, 0, stream>>>(w_qkv, wqkvbf, (4096 * 2048) / 4);
    cvt_bf16_kernel<<<dim3(1024), blk, 0, stream>>>(w_out, woutbf, (2048 * 2048) / 4);

    gemm_nt_bf<<<dim3(32, 32), blk, 0, stream>>>(xbf, wqkvbf, qkv, 4096, 4096, 2048);
    prep_kernel<<<dim3(4096), blk, 0, stream>>>(qkv, cis, qnw, knw, Qb, Kb, Vt);
    attn_kernel<<<dim3(16, 32), blk, 0, stream>>>(Qb, Kb, Vt, xmask, Ob);
    gemm_nt_bf<<<dim3(16, 32), blk, 0, stream>>>(Ob, woutbf, out, 4096, 2048, 2048);
}
